// MessagePassing_74328704025115
// MI455X (gfx1250) — compile-verified, matrix operand order checked by reading, not run on hardware
//
#include <hip/hip_runtime.h>
#include <hip/hip_bf16.h>

#define HIDDEN 128

typedef __attribute__((ext_vector_type(16))) __bf16 v16bf;
typedef __attribute__((ext_vector_type(8)))  __bf16 v8bf;
typedef __attribute__((ext_vector_type(8)))  float  v8f;

// ---------------------------------------------------------------------------
// WMMA bf16 16x16x32 fragment helpers (wave32).
//
// A-matrix (16x32, 16-bit) per ISA layout:
//   lane L (L<16)  holds row M=L,    K = {ko..ko+7, ko+16..ko+23} with ko=0
//   lane L (L>=16) holds row M=L-16, same K pattern with ko=8
//   vector elem i   = K = ko+i       (i=0..7)
//   vector elem 8+i = K = ko+16+i
// B-matrix (32x16) is the mirror with N in place of M, so a B fragment is an
// A-style load from W^T (N-major, K-contiguous).
// C/D (16x16 f32): lane L -> N = L&15; VGPR v -> M = v + 8*(L>>4).
// ---------------------------------------------------------------------------

// A fragment from fp32 row-major source; rowp = &src[(row0 + (lane&15))*128 + kbase]
static __device__ __forceinline__ v16bf load_a_f32(const float* rowp, int lane) {
  const int ko = (lane >> 4) << 3;                 // 0 or 8
  const float4* p = (const float4*)(rowp + ko);    // 16B aligned
  float4 c0 = p[0];                                // K = ko+0..3
  float4 c1 = p[1];                                // K = ko+4..7
  float4 c2 = p[4];                                // K = ko+16..19
  float4 c3 = p[5];                                // K = ko+20..23
  v16bf a;
  a[0]  = (__bf16)c0.x; a[1]  = (__bf16)c0.y; a[2]  = (__bf16)c0.z; a[3]  = (__bf16)c0.w;
  a[4]  = (__bf16)c1.x; a[5]  = (__bf16)c1.y; a[6]  = (__bf16)c1.z; a[7]  = (__bf16)c1.w;
  a[8]  = (__bf16)c2.x; a[9]  = (__bf16)c2.y; a[10] = (__bf16)c2.z; a[11] = (__bf16)c2.w;
  a[12] = (__bf16)c3.x; a[13] = (__bf16)c3.y; a[14] = (__bf16)c3.z; a[15] = (__bf16)c3.w;
  return a;
}

// B fragment from bf16 W^T (N-major, 128 K per row); colp = &WT[n*128 + kbase]
static __device__ __forceinline__ v16bf load_b_bf16(const __bf16* colp, int lane) {
  const int ko = (lane >> 4) << 3;
  v8bf c0 = *(const v8bf*)(colp + ko);        // K = ko..ko+7
  v8bf c1 = *(const v8bf*)(colp + ko + 16);   // K = ko+16..ko+23
  v16bf b;
#pragma unroll
  for (int i = 0; i < 8; ++i) { b[i] = c0[i]; b[8 + i] = c1[i]; }
  return b;
}

static __device__ __forceinline__ v8f wmma_bf16(v16bf a, v16bf b, v8f c) {
  return __builtin_amdgcn_wmma_f32_16x16x32_bf16(
      /*neg_a=*/false, a, /*neg_b=*/false, b,
      /*c_mod=*/(short)0, c, /*reuse_a=*/false, /*reuse_b=*/false);
}

// ---------------------------------------------------------------------------
// Kernel 0: W [384,128] fp32 row-major -> WT[3][128][128] bf16 (N-major)
//           WT[blk][n][k] = bf16(W[(blk*128+k)*128 + n])
// ---------------------------------------------------------------------------
__global__ void prep_w_kernel(const float* __restrict__ W, __bf16* __restrict__ WT) {
  int idx = blockIdx.x * blockDim.x + threadIdx.x;
  const int total = 3 * HIDDEN * HIDDEN;
  if (idx >= total) return;
  int blk = idx / (HIDDEN * HIDDEN);
  int rem = idx - blk * HIDDEN * HIDDEN;
  int n = rem >> 7;
  int k = rem & 127;
  WT[idx] = (__bf16)W[(size_t)(blk * HIDDEN + k) * HIDDEN + n];
}

// ---------------------------------------------------------------------------
// Kernel 1: P1 = node @ W1, P2 = node @ W2  (one wave per 16-row tile)
// ---------------------------------------------------------------------------
__global__ __launch_bounds__(256) void node_proj_kernel(
    const float* __restrict__ nodef, const __bf16* __restrict__ WT,
    float* __restrict__ P1, float* __restrict__ P2, int n_nodes) {
  const int lane = threadIdx.x & 31;
  const int wave = (blockIdx.x * blockDim.x + threadIdx.x) >> 5;
  const int tiles = n_nodes >> 4;
  if (wave >= tiles) return;
  const int row0 = wave << 4;

  const float* rowp = nodef + (size_t)(row0 + (lane & 15)) * HIDDEN;
  v16bf a[4];
#pragma unroll
  for (int kc = 0; kc < 4; ++kc) a[kc] = load_a_f32(rowp + kc * 32, lane);

  const int nlo = lane & 15;
  const int mhi = (lane >> 4) << 3;

#pragma unroll
  for (int h = 0; h < 2; ++h) {
    float* dst = h ? P2 : P1;
    const __bf16* wbase = WT + (size_t)h * HIDDEN * HIDDEN;
#pragma unroll
    for (int nt = 0; nt < 8; ++nt) {
      const __bf16* colp = wbase + (size_t)(nt * 16 + nlo) * HIDDEN;
      v8f acc = {};
#pragma unroll
      for (int kc = 0; kc < 4; ++kc)
        acc = wmma_bf16(a[kc], load_b_bf16(colp + kc * 32, lane), acc);
      const int n = nt * 16 + nlo;
#pragma unroll
      for (int v = 0; v < 8; ++v)
        dst[(size_t)(row0 + v + mhi) * HIDDEN + n] = acc[v];
    }
  }
}

// ---------------------------------------------------------------------------
// Kernel 2: per 16-edge tile: T = edge @ W3, then
//           msg = relu(T + P1[s] + P2[r] + b), atomic scatter-add into out[r]
// ---------------------------------------------------------------------------
__global__ __launch_bounds__(256) void edge_mp_kernel(
    const float* __restrict__ edgef, const int* __restrict__ senders,
    const int* __restrict__ receivers, const __bf16* __restrict__ WT,
    const float* __restrict__ P1, const float* __restrict__ P2,
    const float* __restrict__ bias, float* __restrict__ out, int n_edges) {
  const int lane = threadIdx.x & 31;
  const int wave = (blockIdx.x * blockDim.x + threadIdx.x) >> 5;
  const int tiles = n_edges >> 4;
  if (wave >= tiles) return;
  const int e0 = wave << 4;

  const float* rowp = edgef + (size_t)(e0 + (lane & 15)) * HIDDEN;
  v16bf a[4];
#pragma unroll
  for (int kc = 0; kc < 4; ++kc) a[kc] = load_a_f32(rowp + kc * 32, lane);

  const int nlo = lane & 15;
  const int mhi = (lane >> 4) << 3;

  int sIdx[8], rIdx[8];
#pragma unroll
  for (int v = 0; v < 8; ++v) {
    const int e = e0 + v + mhi;
    sIdx[v] = senders[e];
    rIdx[v] = receivers[e];
  }

  const __bf16* wbase = WT + (size_t)2 * HIDDEN * HIDDEN;
#pragma unroll
  for (int nt = 0; nt < 8; ++nt) {
    const __bf16* colp = wbase + (size_t)(nt * 16 + nlo) * HIDDEN;
    v8f acc = {};
#pragma unroll
    for (int kc = 0; kc < 4; ++kc)
      acc = wmma_bf16(a[kc], load_b_bf16(colp + kc * 32, lane), acc);

    const int n = nt * 16 + nlo;
    const float bn = bias[n];
#pragma unroll
    for (int v = 0; v < 8; ++v) {
      float m = acc[v] + P1[(size_t)sIdx[v] * HIDDEN + n]
                       + P2[(size_t)rIdx[v] * HIDDEN + n] + bn;
      m = fmaxf(m, 0.0f);
      atomicAdd(out + (size_t)rIdx[v] * HIDDEN + n, m);  // global_atomic_add_f32
    }
  }
}

// ---------------------------------------------------------------------------
// Kernel 3: out = LayerNorm(node + aggr), in-place on d_out. One wave per row.
// ---------------------------------------------------------------------------
__global__ __launch_bounds__(256) void layernorm_kernel(
    const float* __restrict__ nodef, const float* __restrict__ lnw,
    const float* __restrict__ lnb, float* __restrict__ out, int n_nodes) {
  const int lane = threadIdx.x & 31;
  const int row = (blockIdx.x * blockDim.x + threadIdx.x) >> 5;
  if (row >= n_nodes) return;

  const float* xr = nodef + (size_t)row * HIDDEN;
  float* ar = out + (size_t)row * HIDDEN;

  float x[4];
  float s = 0.0f;
#pragma unroll
  for (int j = 0; j < 4; ++j) {
    x[j] = xr[lane + 32 * j] + ar[lane + 32 * j];
    s += x[j];
  }
#pragma unroll
  for (int o = 16; o > 0; o >>= 1) s += __shfl_xor(s, o, 32);
  const float mu = s * (1.0f / HIDDEN);

  float v2 = 0.0f;
#pragma unroll
  for (int j = 0; j < 4; ++j) {
    const float d = x[j] - mu;
    v2 += d * d;
  }
#pragma unroll
  for (int o = 16; o > 0; o >>= 1) v2 += __shfl_xor(v2, o, 32);
  const float rs = rsqrtf(v2 * (1.0f / HIDDEN) + 1e-5f);

#pragma unroll
  for (int j = 0; j < 4; ++j) {
    const int c = lane + 32 * j;
    ar[c] = (x[j] - mu) * rs * lnw[c] + lnb[c];
  }
}

// ---------------------------------------------------------------------------
extern "C" void kernel_launch(void* const* d_in, const int* in_sizes, int n_in,
                              void* d_out, int out_size, void* d_ws, size_t ws_size,
                              hipStream_t stream) {
  const float* nodef     = (const float*)d_in[0];
  const int*   senders   = (const int*)d_in[1];
  const int*   receivers = (const int*)d_in[2];
  const float* edgef     = (const float*)d_in[3];
  const float* W         = (const float*)d_in[4];
  const float* bias      = (const float*)d_in[5];
  const float* lnw       = (const float*)d_in[6];
  const float* lnb       = (const float*)d_in[7];
  float* out = (float*)d_out;

  const int n_nodes = in_sizes[0] / HIDDEN;   // 50000
  const int n_edges = in_sizes[1];            // 800000

  // Workspace layout: P1 [N,128] f32 | P2 [N,128] f32 | WT [3*128*128] bf16
  // Requirement: ~51.3 MB.
  float*  P1 = (float*)d_ws;
  float*  P2 = P1 + (size_t)n_nodes * HIDDEN;
  __bf16* WT = (__bf16*)(P2 + (size_t)n_nodes * HIDDEN);

  // d_out doubles as scatter-add accumulator; must be zeroed every call.
  hipMemsetAsync(d_out, 0, (size_t)n_nodes * HIDDEN * sizeof(float), stream);

  prep_w_kernel<<<(3 * HIDDEN * HIDDEN + 255) / 256, 256, 0, stream>>>(W, WT);

  const int node_waves = n_nodes >> 4;   // 3125 tiles of 16 rows
  node_proj_kernel<<<(node_waves * 32 + 255) / 256, 256, 0, stream>>>(
      nodef, WT, P1, P2, n_nodes);

  const int edge_waves = n_edges >> 4;   // 50000 tiles of 16 edges
  edge_mp_kernel<<<(edge_waves * 32 + 255) / 256, 256, 0, stream>>>(
      edgef, senders, receivers, WT, P1, P2, bias, out, n_edges);

  layernorm_kernel<<<(n_nodes * 32 + 255) / 256, 256, 0, stream>>>(
      nodef, lnw, lnb, out, n_nodes);
}
